// TernLinear_14027363189199
// MI455X (gfx1250) — matrix operand order repaired and checked by reading, not compiled, not run on hardware
//
#include <hip/hip_runtime.h>
#include <stdint.h>

// ---- CDNA5 WMMA / TDM vector types ------------------------------------------
typedef __attribute__((ext_vector_type(16))) _Float16 v16h;
typedef __attribute__((ext_vector_type(8)))  _Float16 v8h;
typedef __attribute__((ext_vector_type(4)))  _Float16 v4h;
typedef __attribute__((ext_vector_type(8)))  float    v8f;
typedef __attribute__((ext_vector_type(4)))  unsigned int u32x4;
typedef __attribute__((ext_vector_type(8)))  int      i32x8;
typedef __attribute__((ext_vector_type(4)))  int      i32x4;

// Problem size (fixed by the reference: B = NIP = NOP = 4096)
constexpr int MDIM = 4096;
constexpr int NDIM = 4096;
constexpr int KDIM = 4096;

constexpr int BK   = 32;    // K step (one f16 WMMA depth)
constexpr int LDSK = 40;    // padded K stride in halves (80B rows; TDM pad matches)

// main (TDM) kernel tile: 128(M) x 256(N), 8 waves in 2x4 grid, 64x64 per wave
constexpr int BM  = 128;
constexpr int BNW = 256;
// fallback kernel tile: 128 x 128
constexpr int FBN = 128;

__device__ __forceinline__ _Float16 ternarize(float w) {
    // clip(w,-1,1) -> sign -> zero the band [-0.5, 0.5]; result in {-1,0,+1}
    float r = (w >= 0.0f) ? 1.0f : -1.0f;
    if (w >= -0.5f && w <= 0.5f) r = 0.0f;
    return (_Float16)r;
}

// ---- WMMA fragment loads from LDS --------------------------------------------
// A fragment (16x32 f16, M x K): lane holds row m = lane&15.
// lanes 0-15 : halves 0..7 = K 0..7,  halves 8..15 = K 16..23
// lanes 16-31: halves 0..7 = K 8..15, halves 8..15 = K 24..31
__device__ __forceinline__ v16h frag_a(const _Float16* tile, int row, int lane) {
    const _Float16* p = tile + row * LDSK + ((lane & 16) ? 8 : 0);
    v8h lo = *(const v8h*)(p);        // ds_load_b128
    v8h hi = *(const v8h*)(p + 16);   // ds_load_b128
    v16h f;
#pragma unroll
    for (int i = 0; i < 8; ++i) { f[i] = lo[i]; f[i + 8] = hi[i]; }
    return f;
}

// B fragment (32x16 f16, K x N), stored transposed in LDS as [n][k]:
// lane holds column n = lane&15; lanes 0-15: K 0..7,16..23, lanes 16-31: K 8..15,24..31.
__device__ __forceinline__ v16h frag_b(const _Float16* tile, int row, int lane) {
    const _Float16* p = tile + row * LDSK + ((lane & 16) ? 8 : 0);
    v8h lo = *(const v8h*)(p);        // ds_load_b128
    v8h hi = *(const v8h*)(p + 16);   // ds_load_b128
    v16h f;
#pragma unroll
    for (int i = 0; i < 8; ++i) { f[i] = lo[i]; f[i + 8] = hi[i]; }
    return f;
}

// ---- Tensor Data Mover: DMA a (tile_rows x 32) f16 tile (row stride 4096) ----
// D# group0: count=1 | lds_addr | global_addr | type=2
// D# group1: data_size=2B, pad_enable, pad_interval=16 DWORDs (one 64B row),
//            pad_amount=4 DWORDs (16B) -> LDS row stride = 80B = LDSK halves.
__device__ __forceinline__ void tdm_load_tile(unsigned lds_off, const _Float16* gsrc,
                                              int tile_rows) {
    unsigned long long ga = (unsigned long long)(uintptr_t)gsrc;
    u32x4 g0;
    g0[0] = 1u;                                           // count=1 (valid D#)
    g0[1] = lds_off;                                      // lds_addr (bytes)
    g0[2] = (unsigned)ga;                                 // global_addr[31:0]
    g0[3] = (unsigned)((ga >> 32) & 0x01FFFFFFull)        // global_addr[56:32]
          | (2u << 30);                                   // type=2 ("image")

    i32x8 g1;
    g1[0] = (1 << 16)                                     // data_size = 2 bytes
          | (1 << 20)                                     // pad_enable
          | (3 << 22)                                     // pad_interval: 16 DW
          | (3 << 25);                                    // pad_amount : 4 DW
    g1[1] = (int)(((unsigned)KDIM & 0xFFFFu) << 16);      // tensor_dim0[15:0]
    g1[2] = (int)((((unsigned)KDIM >> 16) & 0xFFFFu)      // tensor_dim0[31:16]
          | (((unsigned)KDIM & 0xFFFFu) << 16));          // tensor_dim1[15:0]
    g1[3] = (int)((((unsigned)KDIM >> 16) & 0xFFFFu)      // tensor_dim1[31:16]
          | ((unsigned)BK << 16));                        // tile_dim0 = 32
    g1[4] = tile_rows & 0xFFFF;                           // tile_dim1, tile_dim2=0
    g1[5] = KDIM;                                         // tensor_dim0_stride[31:0]
    g1[6] = 0;
    g1[7] = 0;

    i32x4 z4 = {0, 0, 0, 0};
#if __has_include(<hip/amd_detail/amd_gfx1250_TDM.h>)
    i32x8 z8 = {0, 0, 0, 0, 0, 0, 0, 0};
    __builtin_amdgcn_tensor_load_to_lds(g0, g1, z4, z4, z8, 0);
#else
    __builtin_amdgcn_tensor_load_to_lds(g0, g1, z4, z4, 0);
#endif
    asm volatile("" ::: "memory");   // TDM writes LDS behind the compiler's back
}

// ---- preprocessing: x fp32 -> f16 --------------------------------------------
__global__ __launch_bounds__(256)
void cvt_x_f16(const float* __restrict__ x, _Float16* __restrict__ xh) {
    size_t i = ((size_t)blockIdx.x * 256 + threadIdx.x) * 4;
    float4 v = *(const float4*)(x + i);
    v4h h;
    h[0] = (_Float16)v.x; h[1] = (_Float16)v.y;
    h[2] = (_Float16)v.z; h[3] = (_Float16)v.w;
    *(v4h*)(xh + i) = h;
}

// ---- preprocessing: w fp32 [k][n] -> ternary f16 transposed [n][k] -----------
__global__ __launch_bounds__(256)
void tern_transpose_w(const float* __restrict__ w, _Float16* __restrict__ wt) {
    __shared__ _Float16 t[32][33];
    const int tx = threadIdx.x & 31;
    const int ty = threadIdx.x >> 5;          // 0..7
    const int k0 = blockIdx.y * 32;
    const int n0 = blockIdx.x * 32;
#pragma unroll
    for (int j = 0; j < 4; ++j) {
        const int k = ty * 4 + j;
        t[k][tx] = ternarize(w[(size_t)(k0 + k) * NDIM + n0 + tx]);  // coalesced in n
    }
    __syncthreads();
#pragma unroll
    for (int j = 0; j < 4; ++j) {
        const int n = ty * 4 + j;
        wt[(size_t)(n0 + n) * KDIM + k0 + tx] = t[tx][n];            // coalesced in k
    }
}

// ---- main GEMM: TDM triple-buffered (depth-2 prefetch), 16 WMMA / wave / step -
__global__ __launch_bounds__(256)
void tern_gemm_wmma_f16_tdm(const _Float16* __restrict__ xh,
                            const _Float16* __restrict__ wt,
                            float* __restrict__ out) {
    __shared__ _Float16 ldsX[3][BM][LDSK];    // x tile,  [m][k]   30 KB
    __shared__ _Float16 ldsW[3][BNW][LDSK];   // wT tile, [n][k]   60 KB

    const int tid  = threadIdx.x;
    const int lane = tid & 31;
    const int wv   = tid >> 5;          // 8 waves
    const int wm   = wv >> 2;           // 0..1  (64 rows each)
    const int wn   = wv & 3;            // 0..3  (64 cols each)

    const int m0 = blockIdx.y * BM;
    const int n0 = blockIdx.x * BNW;

    const _Float16* xbase = xh + (size_t)m0 * KDIM;
    const _Float16* wbase = wt + (size_t)n0 * KDIM;

    v8f acc[4][4];
#pragma unroll
    for (int i = 0; i < 4; ++i)
#pragma unroll
        for (int j = 0; j < 4; ++j) acc[i][j] = (v8f){0,0,0,0,0,0,0,0};

    constexpr int NT = KDIM / BK;       // 128 K steps

    // prologue: TDM-stage tiles 0 and 1 (depth-2 pipeline)
    if (wv == 0) {
        tdm_load_tile((unsigned)(uintptr_t)&ldsX[0][0][0], xbase, BM);
        tdm_load_tile((unsigned)(uintptr_t)&ldsW[0][0][0], wbase, BNW);
        tdm_load_tile((unsigned)(uintptr_t)&ldsX[1][0][0], xbase + (size_t)BK, BM);
        tdm_load_tile((unsigned)(uintptr_t)&ldsW[1][0][0], wbase + (size_t)BK, BNW);
        __builtin_amdgcn_s_wait_tensorcnt(2);   // tile 0 resident (in-order TDM)
    }
    __syncthreads();
    asm volatile("" ::: "memory");

    int cur = 0;
#pragma unroll 1
    for (int kt = 0; kt < NT; ++kt) {
        const int nxt2 = (cur == 2) ? 1 : ((cur == 1) ? 0 : 2);   // (cur+2)%3

        // kick DMA for K-tile kt+2 into the buffer freed at iteration kt-1
        if (kt + 2 < NT && wv == 0) {
            tdm_load_tile((unsigned)(uintptr_t)&ldsX[nxt2][0][0],
                          xbase + (size_t)(kt + 2) * BK, BM);
            tdm_load_tile((unsigned)(uintptr_t)&ldsW[nxt2][0][0],
                          wbase + (size_t)(kt + 2) * BK, BNW);
        }

        const _Float16* Xt = &ldsX[cur][0][0];
        const _Float16* Wt = &ldsW[cur][0][0];

        // 4 resident B fragments, A fragments streamed; 16 WMMAs per wave
        v16h bf[4];
#pragma unroll
        for (int sn = 0; sn < 4; ++sn)
            bf[sn] = frag_b(Wt, wn * 64 + sn * 16 + (lane & 15), lane);

#pragma unroll
        for (int sm = 0; sm < 4; ++sm) {
            v16h af = frag_a(Xt, wm * 64 + sm * 16 + (lane & 15), lane);
#pragma unroll
            for (int sn = 0; sn < 4; ++sn) {
                acc[sm][sn] = __builtin_amdgcn_wmma_f32_16x16x32_f16(
                    /*neg_a=*/false, af, /*neg_b=*/false, bf[sn],
                    /*c_mod=*/(short)0, acc[sm][sn],
                    /*reuse_a=*/false, /*reuse_b=*/false);
            }
        }

        if (kt + 1 < NT) {
            if (wv == 0) {
                if (kt + 2 < NT) __builtin_amdgcn_s_wait_tensorcnt(2); // kt+1 resident
                else             __builtin_amdgcn_s_wait_tensorcnt(0); // drain tail
            }
            __syncthreads();
            asm volatile("" ::: "memory");
        }
        cur = (cur == 2) ? 0 : cur + 1;
    }

    // epilogue: C/D layout: VGPR i -> M = i (+8 for lanes 16-31), N = lane&15
#pragma unroll
    for (int sm = 0; sm < 4; ++sm) {
#pragma unroll
        for (int sn = 0; sn < 4; ++sn) {
            const int bm = m0 + wm * 64 + sm * 16 + ((lane & 16) ? 8 : 0);
            const int bn = n0 + wn * 64 + sn * 16 + (lane & 15);
#pragma unroll
            for (int i = 0; i < 8; ++i)
                out[(size_t)(bm + i) * NDIM + bn] = acc[sm][sn][i];
        }
    }
}

// ---- fallback (no workspace): inline convert/ternarize, 128x128 tile ---------
__global__ __launch_bounds__(256)
void tern_gemm_wmma_f16_inline(const float* __restrict__ x,
                               const float* __restrict__ w,
                               float* __restrict__ out) {
    __shared__ _Float16 ldsX[2][BM][LDSK];
    __shared__ _Float16 ldsW[2][FBN][LDSK];

    const int tid  = threadIdx.x;
    const int lane = tid & 31;
    const int wv   = tid >> 5;
    const int wm   = wv >> 2;
    const int wn   = wv & 3;
    const int m0 = blockIdx.y * BM;
    const int n0 = blockIdx.x * FBN;

    v8f acc[4][2];
#pragma unroll
    for (int i = 0; i < 4; ++i)
#pragma unroll
        for (int j = 0; j < 2; ++j) acc[i][j] = (v8f){0,0,0,0,0,0,0,0};

    auto fetch_tiles = [&](int k0, float4 (&xr)[4], float4 (&wr)[4]) {
#pragma unroll
        for (int p = 0; p < 4; ++p) {
            const int id = p * 256 + tid;
            const int xm = id >> 3, xk = (id & 7) * 4;
            xr[p] = *(const float4*)(x + (size_t)(m0 + xm) * KDIM + k0 + xk);
            const int wk = id >> 5, wn4 = (id & 31) * 4;
            wr[p] = *(const float4*)(w + (size_t)(k0 + wk) * NDIM + n0 + wn4);
        }
    };
    auto stage_tiles = [&](int buf, const float4 (&xr)[4], const float4 (&wr)[4]) {
#pragma unroll
        for (int p = 0; p < 4; ++p) {
            const int id = p * 256 + tid;
            const int xm = id >> 3, xk = (id & 7) * 4;
            v4h hx;
            hx[0] = (_Float16)xr[p].x; hx[1] = (_Float16)xr[p].y;
            hx[2] = (_Float16)xr[p].z; hx[3] = (_Float16)xr[p].w;
            *(v4h*)&ldsX[buf][xm][xk] = hx;
            const int wk = id >> 5, wn4 = (id & 31) * 4;
            ldsW[buf][wn4 + 0][wk] = ternarize(wr[p].x);
            ldsW[buf][wn4 + 1][wk] = ternarize(wr[p].y);
            ldsW[buf][wn4 + 2][wk] = ternarize(wr[p].z);
            ldsW[buf][wn4 + 3][wk] = ternarize(wr[p].w);
        }
    };

    {
        float4 xr[4], wr[4];
        fetch_tiles(0, xr, wr);
        stage_tiles(0, xr, wr);
    }
    __syncthreads();

    constexpr int NT = KDIM / BK;
    for (int kt = 0; kt < NT; ++kt) {
        const int cur = kt & 1;
        float4 xr[4], wr[4];
        const bool more = (kt + 1 < NT);
        if (more) fetch_tiles((kt + 1) * BK, xr, wr);

        const _Float16* Xt = &ldsX[cur][0][0];
        const _Float16* Wt = &ldsW[cur][0][0];

        v16h bf[2];
#pragma unroll
        for (int sn = 0; sn < 2; ++sn)
            bf[sn] = frag_b(Wt, wn * 32 + sn * 16 + (lane & 15), lane);
#pragma unroll
        for (int sm = 0; sm < 4; ++sm) {
            v16h af = frag_a(Xt, wm * 64 + sm * 16 + (lane & 15), lane);
#pragma unroll
            for (int sn = 0; sn < 2; ++sn) {
                acc[sm][sn] = __builtin_amdgcn_wmma_f32_16x16x32_f16(
                    false, af, false, bf[sn], (short)0, acc[sm][sn], false, false);
            }
        }

        if (more) {
            __syncthreads();
            stage_tiles(cur ^ 1, xr, wr);
            __syncthreads();
        }
    }

#pragma unroll
    for (int sm = 0; sm < 4; ++sm) {
#pragma unroll
        for (int sn = 0; sn < 2; ++sn) {
            const int bm = m0 + wm * 64 + sm * 16 + ((lane & 16) ? 8 : 0);
            const int bn = n0 + wn * 32 + sn * 16 + (lane & 15);
#pragma unroll
            for (int i = 0; i < 8; ++i)
                out[(size_t)(bm + i) * NDIM + bn] = acc[sm][sn][i];
        }
    }
}

// ---- host launcher -----------------------------------------------------------
extern "C" void kernel_launch(void* const* d_in, const int* in_sizes, int n_in,
                              void* d_out, int out_size, void* d_ws, size_t ws_size,
                              hipStream_t stream) {
    (void)in_sizes; (void)n_in; (void)out_size;
    const float* x = (const float*)d_in[0];      // (4096, 4096) fp32
    const float* w = (const float*)d_in[1];      // (4096, 4096) fp32
    float* out = (float*)d_out;                  // (4096, 4096) fp32

    const size_t need = (size_t)MDIM * KDIM * sizeof(_Float16)
                      + (size_t)NDIM * KDIM * sizeof(_Float16);   // 64 MB

    if (ws_size >= need) {
        _Float16* xh = (_Float16*)d_ws;
        _Float16* wt = xh + (size_t)MDIM * KDIM;
        cvt_x_f16<<<(MDIM * (size_t)KDIM) / (4 * 256), 256, 0, stream>>>(x, xh);
        tern_transpose_w<<<dim3(NDIM / 32, KDIM / 32), 256, 0, stream>>>(w, wt);
        tern_gemm_wmma_f16_tdm<<<dim3(NDIM / BNW, MDIM / BM), 256, 0, stream>>>(xh, wt, out);
    } else {
        tern_gemm_wmma_f16_inline<<<dim3(NDIM / FBN, MDIM / BM), 256, 0, stream>>>(x, w, out);
    }
}
